// CausalVarlenSelfAttention_76235669504654
// MI455X (gfx1250) — matrix operand order repaired and checked by reading, not compiled, not run on hardware
//
#include <hip/hip_runtime.h>
#include <hip/hip_bf16.h>

typedef __attribute__((ext_vector_type(16))) _Float16 v16h;
typedef __attribute__((ext_vector_type(8)))  _Float16 v8h;
typedef __attribute__((ext_vector_type(8)))  float    v8f;

#define WMMA_F32_F16(a,b,c) \
  __builtin_amdgcn_wmma_f32_16x16x32_f16(false,(a),false,(b),(short)0,(c),false,false)

__device__ __forceinline__ v16h pack16(v8h c0, v8h c1){
  v16h r;
#pragma unroll
  for (int i=0;i<8;++i){ r[i]=c0[i]; r[i+8]=c1[i]; }
  return r;
}

// A fragment (16x32 f16, M x K), source row-major [row][ld], row = M index.
// Lanes 0-15: K 0-7 & 16-23 ; lanes 16-31: K 8-15 & 24-31 (ISA 7.12.2 layout).
__device__ __forceinline__ v16h load_a_frag(const _Float16* p, int row0, int k0, int ld){
  int l = threadIdx.x & 31;
  const _Float16* q = p + (size_t)(row0 + (l & 15)) * ld + (size_t)(k0 + ((l >> 4) * 8));
  v8h c0 = *(const v8h*)q;
  v8h c1 = *(const v8h*)(q + 16);
  return pack16(c0, c1);
}

// B fragment (32x16 f16, K x N) where B[k][n] = src[col0+n][k] (src row-major [n][ld]).
// Lanes 0-15: K 0-7 & 16-23 ; lanes 16-31: K 8-15 & 24-31 (same as A fragment).
__device__ __forceinline__ v16h load_b_frag(const _Float16* p, int col0, int k0, int ld){
  int l = threadIdx.x & 31;
  const _Float16* q = p + (size_t)(col0 + (l & 15)) * ld + (size_t)(k0 + ((l >> 4) * 8));
  v8h c0 = *(const v8h*)q;
  v8h c1 = *(const v8h*)(q + 16);
  return pack16(c0, c1);
}

// ---------------- fp32 -> f16 conversion ----------------
__global__ void cvt_kernel(const float* __restrict__ src, _Float16* __restrict__ dst, long n){
  long i = (long)blockIdx.x * blockDim.x + threadIdx.x;
  if (i < n) dst[i] = (_Float16)src[i];
}

// ---------------- generic C = A * B^T (f16 in, f32 out) ----------------
// A: [M][K] row-major f16 ; Bm: [N][K] row-major f16 ; C: [M][N] f32.
// One wave computes a 32x64 output tile: two A fragments share every B fragment,
// giving 8 WMMAs per 32-wide K step for 12 b128 loads.
__global__ void gemm_wmma_f16(const _Float16* __restrict__ A,
                              const _Float16* __restrict__ Bm,
                              float* __restrict__ C,
                              int M, int N, int K){
  int wid = (int)((blockIdx.x * blockDim.x + threadIdx.x) >> 5);
  int n64 = N >> 6;
  int total = (M >> 5) * n64;
  if (wid >= total) return;
  int mt = wid / n64, nt = wid % n64;
  int row0 = mt << 5, col0 = nt << 6;
  v8f acc0[4] = {};
  v8f acc1[4] = {};
  for (int k = 0; k < K; k += 32){
    // prefetch next K-step of A rows (global_prefetch_b8; cheap, no counter)
    if (k + 32 < K){
      int l = threadIdx.x & 31;
      __builtin_prefetch(A + (size_t)(row0 + l) * K + (k + 32), 0, 1);
    }
    v16h af0 = load_a_frag(A, row0,      k, K);
    v16h af1 = load_a_frag(A, row0 + 16, k, K);
#pragma unroll
    for (int t = 0; t < 4; ++t){
      v16h bf = load_b_frag(Bm, col0 + t*16, k, K);
      acc0[t] = WMMA_F32_F16(af0, bf, acc0[t]);
      acc1[t] = WMMA_F32_F16(af1, bf, acc1[t]);
    }
  }
  int ln = threadIdx.x & 15, hf = (threadIdx.x >> 4) & 1;
#pragma unroll
  for (int t = 0; t < 4; ++t)
#pragma unroll
    for (int i = 0; i < 8; ++i){
      C[(size_t)(row0 + i + 8*hf)      * N + (size_t)(col0 + t*16 + ln)] = acc0[t][i];
      C[(size_t)(row0 + 16 + i + 8*hf) * N + (size_t)(col0 + t*16 + ln)] = acc1[t][i];
    }
}

// ---------------- rotary + rms-norm + layout ----------------
// qkv: [N][3][16][64] f32. Writes Qh/Kh [B][H][S][64] f16 (Q pre-scaled by 1/8),
// Vt [B][H][64][S] f16 (transposed so P@V B-fragments are contiguous).
__global__ void rope_rms_kernel(const float* __restrict__ qkv,
                                const float* __restrict__ cosc,
                                const float* __restrict__ sinc,
                                const int* __restrict__ pid,
                                _Float16* __restrict__ Qh,
                                _Float16* __restrict__ Kh,
                                _Float16* __restrict__ Vt,
                                int Ntok, int S){
  int wid = (int)((blockIdx.x * blockDim.x + threadIdx.x) >> 5);
  int t = wid >> 4;
  if (t >= Ntok) return;
  int h = wid & 15;
  int d = threadIdx.x & 31;
  int b = t / S, s = t - b * S;
  int bh = b * 16 + h;
  int pos = pid[t];
  float cv = cosc[pos * 32 + d];
  float sv = sinc[pos * 32 + d];
  const float* base = qkv + (size_t)t * 3072 + h * 64;
  // Q: rotary -> rms -> *1/sqrt(64)
  {
    float x1 = base[d], x2 = base[d + 32];
    float y1 =  x1 * cv + x2 * sv;
    float y2 = -x1 * sv + x2 * cv;
    float ss = y1 * y1 + y2 * y2;
#pragma unroll
    for (int m = 1; m < 32; m <<= 1) ss += __shfl_xor(ss, m, 32);
    float r = rsqrtf(ss * (1.0f / 64.0f) + 1.1920929e-07f) * 0.125f;
    _Float16* dst = Qh + ((size_t)bh * S + s) * 64;
    dst[d] = (_Float16)(y1 * r); dst[d + 32] = (_Float16)(y2 * r);
  }
  // K: rotary -> rms
  {
    const float* kb2 = base + 1024;
    float x1 = kb2[d], x2 = kb2[d + 32];
    float y1 =  x1 * cv + x2 * sv;
    float y2 = -x1 * sv + x2 * cv;
    float ss = y1 * y1 + y2 * y2;
#pragma unroll
    for (int m = 1; m < 32; m <<= 1) ss += __shfl_xor(ss, m, 32);
    float r = rsqrtf(ss * (1.0f / 64.0f) + 1.1920929e-07f);
    _Float16* dst = Kh + ((size_t)bh * S + s) * 64;
    dst[d] = (_Float16)(y1 * r); dst[d + 32] = (_Float16)(y2 * r);
  }
  // V: transpose into [dim][seq]
  {
    const float* vb2 = base + 2048;
    _Float16* dst = Vt + (size_t)bh * 64 * S + s;
    dst[(size_t)d * S]        = (_Float16)vb2[d];
    dst[(size_t)(d + 32) * S] = (_Float16)vb2[d + 32];
  }
}

// ---------------- flash attention: one wave per (b,h,16-query tile) ----------------
__global__ void attn_kernel(const _Float16* __restrict__ Qh,
                            const _Float16* __restrict__ Kh,
                            const _Float16* __restrict__ Vt,
                            _Float16* __restrict__ Ao,
                            int S){
  __shared__ __align__(16) _Float16 pbuf[16 * 32];
  int qt_per = S >> 4;
  int qt = blockIdx.x % qt_per;
  int bh = blockIdx.x / qt_per;
  int h = bh & 15, b = bh >> 4;
  int qb = qt << 4;
  int l = threadIdx.x, ln = l & 15, hf = l >> 4;

  const _Float16* Qb = Qh + (size_t)bh * S * 64;
  const _Float16* Kb = Kh + (size_t)bh * S * 64;
  const _Float16* Vb = Vt + (size_t)bh * 64 * S;

  v16h qf0 = load_a_frag(Qb, qb, 0, 64);
  v16h qf1 = load_a_frag(Qb, qb, 32, 64);

  v8f o[4] = {};
  float mrow[8], lrow[8];
#pragma unroll
  for (int i = 0; i < 8; ++i){ mrow[i] = -3.0e38f; lrow[i] = 0.0f; }

  int nkb = ((qb + 15) >> 5) + 1;
  for (int kb = 0; kb < nkb; ++kb){
    int key0 = kb << 5;
    // scores: two 16x16 tiles (keys key0..+15, key0+16..+31), K dim = head_dim 64
    v16h kf00 = load_b_frag(Kb, key0,      0, 64);
    v16h kf01 = load_b_frag(Kb, key0,     32, 64);
    v16h kf10 = load_b_frag(Kb, key0 + 16, 0, 64);
    v16h kf11 = load_b_frag(Kb, key0 + 16, 32, 64);
    v8f s0 = {}, s1 = {};
    s0 = WMMA_F32_F16(qf0, kf00, s0);
    s0 = WMMA_F32_F16(qf1, kf01, s0);
    s1 = WMMA_F32_F16(qf0, kf10, s1);
    s1 = WMMA_F32_F16(qf1, kf11, s1);
    // causal mask (only diagonal blocks need it)
    if (key0 + 31 > qb){
      int ka = key0 + ln, kc = ka + 16;
#pragma unroll
      for (int i = 0; i < 8; ++i){
        int qr = qb + i + 8 * hf;
        s0[i] = (ka <= qr) ? s0[i] : -1.0e30f;
        s1[i] = (kc <= qr) ? s1[i] : -1.0e30f;
      }
    }
    // online softmax: row M=i lives in VGPR i across a 16-lane half
#pragma unroll
    for (int i = 0; i < 8; ++i){
      float mx = fmaxf(s0[i], s1[i]);
#pragma unroll
      for (int m = 1; m < 16; m <<= 1) mx = fmaxf(mx, __shfl_xor(mx, m, 32));
      float mn = fmaxf(mrow[i], mx);
      float al = __expf(mrow[i] - mn);
      mrow[i] = mn;
      float p0 = __expf(s0[i] - mn);
      float p1 = __expf(s1[i] - mn);
      float ps = p0 + p1;
#pragma unroll
      for (int m = 1; m < 16; m <<= 1) ps += __shfl_xor(ps, m, 32);
      lrow[i] = lrow[i] * al + ps;
#pragma unroll
      for (int t = 0; t < 4; ++t) o[t][i] *= al;
      pbuf[(i + 8 * hf) * 32 + ln]      = (_Float16)p0;
      pbuf[(i + 8 * hf) * 32 + ln + 16] = (_Float16)p1;
    }
    __syncthreads(); // single-wave WG: compiler fence + dscnt wait
    // relayout P (C-fragment) -> A-fragment via LDS
    v8h c0 = *(const v8h*)&pbuf[ln * 32 + hf * 8];
    v8h c1 = *(const v8h*)&pbuf[ln * 32 + hf * 8 + 16];
    v16h pf = pack16(c0, c1);
    __syncthreads();
    // O += P @ V  (V transposed: dim rows contiguous along keys)
#pragma unroll
    for (int t = 0; t < 4; ++t){
      v16h vf = load_b_frag(Vb, t * 16, key0, S);
      o[t] = WMMA_F32_F16(pf, vf, o[t]);
    }
  }
  // normalize + write token-major [N][1024] f16
#pragma unroll
  for (int i = 0; i < 8; ++i){
    float inv = 1.0f / lrow[i];
    size_t row = (size_t)(b * S + qb + i + 8 * hf);
#pragma unroll
    for (int t = 0; t < 4; ++t)
      Ao[row * 1024 + (size_t)(h * 64 + t * 16 + ln)] = (_Float16)(o[t][i] * inv);
  }
}

extern "C" void kernel_launch(void* const* d_in, const int* in_sizes, int n_in,
                              void* d_out, int out_size, void* d_ws, size_t ws_size,
                              hipStream_t stream){
  const float* x    = (const float*)d_in[0];
  const float* Wqkv = (const float*)d_in[1];
  const float* Wo   = (const float*)d_in[2];
  const float* cosc = (const float*)d_in[3];
  const float* sinc = (const float*)d_in[4];
  const int*   pid  = (const int*)d_in[6];
  float* out = (float*)d_out;

  const int E = 1024;
  int N = in_sizes[0] / E;     // tokens
  int S = in_sizes[3] / 32;    // seqlen (cos cache is [S][32])
  int B = N / S;

  char* ws = (char*)d_ws;
  size_t off = 0;
  auto carve = [&](size_t bytes) -> char* {
    char* p = ws + off;
    off += (bytes + 255) & ~(size_t)255;
    return p;
  };
  _Float16* x16   = (_Float16*)carve((size_t)N * E * 2);
  _Float16* w16   = (_Float16*)carve((size_t)3 * E * E * 2);
  _Float16* wo16  = (_Float16*)carve((size_t)E * E * 2);
  float*    qkv32 = (float*)   carve((size_t)N * 3 * E * 4);
  _Float16* Qh    = (_Float16*)carve((size_t)N * E * 2);
  _Float16* Kh    = (_Float16*)carve((size_t)N * E * 2);
  _Float16* Vt    = (_Float16*)carve((size_t)N * E * 2);
  _Float16* Ao    = (_Float16*)carve((size_t)N * E * 2);

  { // f32 -> f16 conversions
    long n = (long)N * E;
    cvt_kernel<<<(int)((n + 255) / 256), 256, 0, stream>>>(x, x16, n);
    n = (long)3 * E * E;
    cvt_kernel<<<(int)((n + 255) / 256), 256, 0, stream>>>(Wqkv, w16, n);
    n = (long)E * E;
    cvt_kernel<<<(int)((n + 255) / 256), 256, 0, stream>>>(Wo, wo16, n);
  }
  { // qkv = x @ Wqkv^T   (32-row tiles)
    int total = (N / 32) * ((3 * E) / 64);
    gemm_wmma_f16<<<(total * 32 + 255) / 256, 256, 0, stream>>>(x16, w16, qkv32, N, 3 * E, E);
  }
  { // rotary + rms + relayout
    int waves = N * 16;
    rope_rms_kernel<<<(waves * 32 + 255) / 256, 256, 0, stream>>>(qkv32, cosc, sinc, pid,
                                                                  Qh, Kh, Vt, N, S);
  }
  { // flash attention
    int blocks = B * 16 * (S / 16);
    attn_kernel<<<blocks, 32, 0, stream>>>(Qh, Kh, Vt, Ao, S);
  }
  { // out = attn @ Wo^T  (fp32 output)
    int total = (N / 32) * (E / 64);
    gemm_wmma_f16<<<(total * 32 + 255) / 256, 256, 0, stream>>>(Ao, wo16, out, N, E, E);
  }
}
